// topoGraph_88562225643607
// MI455X (gfx1250) — compile-verified, matrix operand order checked by reading, not run on hardware
//
#include <hip/hip_runtime.h>
#include <hip/hip_bf16.h>

typedef __attribute__((ext_vector_type(16))) _Float16 v16h;
typedef __attribute__((ext_vector_type(8)))  _Float16 v8h;
typedef __attribute__((ext_vector_type(8)))  float    v8f;

#define F_IN 256
#define C    128

// ---------------------------------------------------------------------------
// Fragment loaders for V_WMMA_F32_16X16X32_F16.
// 16-bit A 16x32 (and by symmetry B 32x16 when B is given as rows of Bt,
// i.e. W[n,k] row-major): lane L in [0,16): row = L, K = 0..7 in elems 0..7,
// K = 16..23 in elems 8..15. Lane L in [16,32): row = L-16, K = 8..15 and
// K = 24..31. So: row = lane&15, koff = (lane>>4)*8, load 8 halves at
// base+row*ld+koff and 8 halves at +16.
// ---------------------------------------------------------------------------

// A-fragment from row-major f32 (x is consumed exactly once -> convert here)
__device__ inline v16h frag_from_f32(const float* __restrict__ tile, int ld, int lane) {
    const int row  = lane & 15;
    const int koff = (lane >> 4) << 3;                  // 0 or 8
    const float* p0 = tile + row * ld + koff;
    const float4 a0 = *(const float4*)(p0);
    const float4 a1 = *(const float4*)(p0 + 4);
    const float4 b0 = *(const float4*)(p0 + 16);
    const float4 b1 = *(const float4*)(p0 + 20);
    v16h r;
    r[0]  = (_Float16)a0.x; r[1]  = (_Float16)a0.y; r[2]  = (_Float16)a0.z; r[3]  = (_Float16)a0.w;
    r[4]  = (_Float16)a1.x; r[5]  = (_Float16)a1.y; r[6]  = (_Float16)a1.z; r[7]  = (_Float16)a1.w;
    r[8]  = (_Float16)b0.x; r[9]  = (_Float16)b0.y; r[10] = (_Float16)b0.z; r[11] = (_Float16)b0.w;
    r[12] = (_Float16)b1.x; r[13] = (_Float16)b1.y; r[14] = (_Float16)b1.z; r[15] = (_Float16)b1.w;
    return r;
}

// B-fragment from row-major, pre-converted f16 weights (two b128 loads)
__device__ inline v16h frag_from_f16(const _Float16* __restrict__ tile, int ld, int lane) {
    const int row  = lane & 15;
    const int koff = (lane >> 4) << 3;
    const _Float16* p0 = tile + row * ld + koff;
    const v8h a = *(const v8h*)(p0);
    const v8h b = *(const v8h*)(p0 + 16);
    v16h r;
#pragma unroll
    for (int i = 0; i < 8; ++i) { r[i] = a[i]; r[8 + i] = b[i]; }
    return r;
}

// A-fragment from LDS f16 (h tile staging)
__device__ inline v16h frag_from_lds(const _Float16* tile, int ld, int lane) {
    const int row  = lane & 15;
    const int koff = (lane >> 4) << 3;
    const _Float16* p0 = tile + row * ld + koff;
    const v8h a = *(const v8h*)(p0);
    const v8h b = *(const v8h*)(p0 + 16);
    v16h r;
#pragma unroll
    for (int i = 0; i < 8; ++i) { r[i] = a[i]; r[8 + i] = b[i]; }
    return r;
}

// ---------------------------------------------------------------------------
// One-shot f32 -> f16 weight conversion (vectorized float4 -> 4 halves)
// ---------------------------------------------------------------------------
__global__ void topo_cvt_weights(const float* __restrict__ in,
                                 _Float16* __restrict__ out, int n4) {
    int i = blockIdx.x * blockDim.x + threadIdx.x;
    if (i >= n4) return;
    const float4 v = *(const float4*)(in + (size_t)i * 4);
    v8h* dst = (v8h*)out;                               // write 8B chunks
    _Float16 h[4] = { (_Float16)v.x, (_Float16)v.y, (_Float16)v.z, (_Float16)v.w };
    *(uint2*)(out + (size_t)i * 4) = *(const uint2*)h;
    (void)dst;
}

// ---------------------------------------------------------------------------
// Fused GEMM: z = relu(x @ W1^T + b1) @ W2^T
// One wave per 16-row M tile; 8 waves (256 threads) per block.
// ---------------------------------------------------------------------------
__global__ __launch_bounds__(256)
void topo_gemm_fused(const float* __restrict__ x,        // [N, 256] f32
                     const _Float16* __restrict__ W1h,   // [128, 256] f16
                     const float* __restrict__ b1,       // [128]
                     const _Float16* __restrict__ W2h,   // [128, 128] f16
                     float* __restrict__ z,              // [N, 128] (ws)
                     int N) {
    __shared__ _Float16 sH[8 * 16 * C];                  // per-wave 16x128 f16 h tile

    const int lane = threadIdx.x & 31;
    const int wave = threadIdx.x >> 5;
    const int row0 = (blockIdx.x * 8 + wave) << 4;       // 16-row tile start
    if (row0 >= N) return;

    _Float16* myH = sH + wave * (16 * C);

    // ---- GEMM 1: h = relu(x @ W1^T + b1), h is 16 x 128 -------------------
    v8f acc[8];
#pragma unroll
    for (int n = 0; n < 8; ++n) acc[n] = (v8f){};

#pragma unroll
    for (int ks = 0; ks < F_IN / 32; ++ks) {             // 8 K-steps of 32
        const v16h afrag = frag_from_f32(x + (size_t)row0 * F_IN + ks * 32, F_IN, lane);
#pragma unroll
        for (int n = 0; n < 8; ++n) {                    // 8 N tiles of 16
            const v16h bfrag = frag_from_f16(W1h + (size_t)(n * 16) * F_IN + ks * 32, F_IN, lane);
            acc[n] = __builtin_amdgcn_wmma_f32_16x16x32_f16(
                false, afrag, false, bfrag, (short)0, acc[n], false, false);
        }
    }

    // bias + relu, convert to f16, stage to LDS in row-major [16][128]
    const int mhi  = (lane >> 4) << 3;                   // 0 or 8
    const int ncol = lane & 15;
#pragma unroll
    for (int n = 0; n < 8; ++n) {
        const float bias = b1[n * 16 + ncol];
#pragma unroll
        for (int i = 0; i < 8; ++i) {
            float v = acc[n][i] + bias;
            v = v > 0.0f ? v : 0.0f;
            myH[(i + mhi) * C + (n * 16 + ncol)] = (_Float16)v;
        }
    }
    // same-wave LDS RAW: DS ops are in-order within a wave; compiler inserts
    // the s_wait_dscnt before the dependent ds_load.

    // ---- GEMM 2: z = h @ W2^T, 16 x 128 ----------------------------------
    v8f acc2[8];
#pragma unroll
    for (int n = 0; n < 8; ++n) acc2[n] = (v8f){};

#pragma unroll
    for (int ks = 0; ks < C / 32; ++ks) {                // 4 K-steps of 32
        const v16h afrag = frag_from_lds(myH + ks * 32, C, lane);
#pragma unroll
        for (int n = 0; n < 8; ++n) {
            const v16h bfrag = frag_from_f16(W2h + (size_t)(n * 16) * C + ks * 32, C, lane);
            acc2[n] = __builtin_amdgcn_wmma_f32_16x16x32_f16(
                false, afrag, false, bfrag, (short)0, acc2[n], false, false);
        }
    }

    // write z tile (f32) to workspace
#pragma unroll
    for (int n = 0; n < 8; ++n) {
#pragma unroll
        for (int i = 0; i < 8; ++i) {
            const int r = row0 + i + mhi;
            z[(size_t)r * C + (n * 16 + ncol)] = acc2[n][i];
        }
    }
}

// ---------------------------------------------------------------------------
// Degree / normalization kernels
// ---------------------------------------------------------------------------
__global__ void topo_deg_init(float* deg, int N) {
    int i = blockIdx.x * blockDim.x + threadIdx.x;
    if (i < N) deg[i] = 1.0f;                            // self-loop contribution
}

__global__ void topo_deg_accum(const int* __restrict__ dst, float* deg, int E) {
    int i = blockIdx.x * blockDim.x + threadIdx.x;
    if (i < E) atomicAdd(&deg[dst[i]], 1.0f);
}

__global__ void topo_dinv(float* deg, int N) {
    int i = blockIdx.x * blockDim.x + threadIdx.x;
    if (i < N) deg[i] = rsqrtf(deg[i]);                  // deg >= 1 always
}

// out[i,:] = b2 + dinv[i]^2 * z[i,:]   (bias + self-loop edge)
__global__ void topo_out_init(const float* __restrict__ z,
                              const float* __restrict__ dinv,
                              const float* __restrict__ b2,
                              float* __restrict__ out, int N) {
    int t = blockIdx.x * blockDim.x + threadIdx.x;       // one float4 per thread
    int node = t >> 5;
    int q    = (t & 31) << 2;
    if (node >= N) return;
    float di = dinv[node];
    float nm = di * di;
    const float4 zv = *(const float4*)(z + (size_t)node * C + q);
    const float4 bv = *(const float4*)(b2 + q);
    float4 o;
    o.x = bv.x + nm * zv.x;
    o.y = bv.y + nm * zv.y;
    o.z = bv.z + nm * zv.z;
    o.w = bv.w + nm * zv.w;
    *(float4*)(out + (size_t)node * C + q) = o;
}

// one wave per edge: lane handles channels [4*lane, 4*lane+3]
__global__ __launch_bounds__(256)
void topo_scatter(const int* __restrict__ src, const int* __restrict__ dst,
                  const float* __restrict__ z, const float* __restrict__ dinv,
                  float* __restrict__ out, int E) {
    int e    = (blockIdx.x * blockDim.x + threadIdx.x) >> 5;
    int lane = threadIdx.x & 31;
    if (e >= E) return;
    int s = src[e];
    int d = dst[e];
    float nm = dinv[s] * dinv[d];
    const float4 zv = *(const float4*)(z + (size_t)s * C + (lane << 2));
    float* o = out + (size_t)d * C + (lane << 2);
    atomicAdd(o + 0, nm * zv.x);
    atomicAdd(o + 1, nm * zv.y);
    atomicAdd(o + 2, nm * zv.z);
    atomicAdd(o + 3, nm * zv.w);
}

// ---------------------------------------------------------------------------
extern "C" void kernel_launch(void* const* d_in, const int* in_sizes, int n_in,
                              void* d_out, int out_size, void* d_ws, size_t ws_size,
                              hipStream_t stream) {
    const float* x  = (const float*)d_in[0];
    const int*   ei = (const int*)d_in[1];
    const float* W1 = (const float*)d_in[2];
    const float* b1 = (const float*)d_in[3];
    const float* W2 = (const float*)d_in[4];
    const float* b2 = (const float*)d_in[5];
    float* out = (float*)d_out;

    const int N = in_sizes[0] / F_IN;
    const int E = in_sizes[1] / 2;
    const int* src = ei;
    const int* dst = ei + E;

    // workspace layout: z [N*C] f32 | deg/dinv [N] f32 | W1h [C*F_IN] f16 | W2h [C*C] f16
    float*      z   = (float*)d_ws;
    float*      deg = z + (size_t)N * C;
    _Float16*   W1h = (_Float16*)(deg + N);
    _Float16*   W2h = W1h + (size_t)C * F_IN;

    // one-shot weight conversion to f16
    topo_cvt_weights<<<((C * F_IN / 4) + 255) / 256, 256, 0, stream>>>(W1, W1h, C * F_IN / 4);
    topo_cvt_weights<<<((C * C    / 4) + 255) / 256, 256, 0, stream>>>(W2, W2h, C * C / 4);

    topo_deg_init<<<(N + 255) / 256, 256, 0, stream>>>(deg, N);
    topo_deg_accum<<<(E + 255) / 256, 256, 0, stream>>>(dst, deg, E);
    topo_dinv<<<(N + 255) / 256, 256, 0, stream>>>(deg, N);

    const int mtiles  = (N + 15) / 16;
    const int gblocks = (mtiles + 7) / 8;                // 8 waves per block
    topo_gemm_fused<<<gblocks, 256, 0, stream>>>(x, W1h, b1, W2h, z, N);

    const long long initThreads = (long long)N * 32;     // one float4 per thread
    topo_out_init<<<(int)((initThreads + 255) / 256), 256, 0, stream>>>(z, deg, b2, out, N);

    topo_scatter<<<(E + 7) / 8, 256, 0, stream>>>(src, dst, z, deg, out, E);
}
